// LSTM_65867618451924
// MI455X (gfx1250) — compile-verified, matrix operand order checked by reading, not run on hardware
//
#include <hip/hip_runtime.h>
#include <hip/hip_bf16.h>

// ---------------------------------------------------------------------------
// LSTM (cell-state recurrent variant) for MI455X / gfx1250.
//   pre = X[t] @ Wx^T + b + c_{t-1} @ Wh^T     (fused K = IN + H per step)
//   i,f,o,g gates -> c,h update with per-(t,b) mask.
// Matrix work runs on v_wmma_f32_16x16x32_bf16 (fp32 accumulate) with a
// 2-deep software pipeline so L2 load latency overlaps the WMMA stream.
// ---------------------------------------------------------------------------

typedef __bf16 bf16_t;
typedef __attribute__((ext_vector_type(16))) __bf16 v16bf;
typedef __attribute__((ext_vector_type(8)))  __bf16 v8bf;
typedef __attribute__((ext_vector_type(8)))  float  v8f;

#define TT  512
#define BB  64
#define INN 1024
#define HH  1024

union FragU { v16bf v; v8bf h[2]; };

// 16-bit A/B fragment for 16x16x32: per lane, 16 bf16 = {k..k+7} U {k+16..k+23}
// starting at (row_base + koff). Two 16-byte contiguous loads.
__device__ __forceinline__ v16bf load_frag(const bf16_t* __restrict__ p) {
    FragU u;
    u.h[0] = *(const v8bf*)(p);
    u.h[1] = *(const v8bf*)(p + 16);
    return u.v;
}

__device__ __forceinline__ v8f wmma_bf16(v16bf a, v16bf b, v8f c) {
    return __builtin_amdgcn_wmma_f32_16x16x32_bf16(
        /*neg_a=*/false, a, /*neg_b=*/false, b,
        /*c_mod=*/(short)0, c, /*reuse_a=*/false, /*reuse_b=*/false);
}

__device__ __forceinline__ float sigmoidf_fast(float x) {
    return 1.0f / (1.0f + __expf(-x));
}

// 2-deep software-pipelined GEMM phase: accumulate A(16xK) x B(Kx16) for the
// four gate columns into acc0..acc3. K must be a multiple of 32 and >= 64.
template <int K>
__device__ __forceinline__ void gemm_phase(
    const bf16_t* __restrict__ aP,
    const bf16_t* __restrict__ g0, const bf16_t* __restrict__ g1,
    const bf16_t* __restrict__ g2, const bf16_t* __restrict__ g3,
    v8f& acc0, v8f& acc1, v8f& acc2, v8f& acc3)
{
    // stage 0 and stage 1 prefetch
    v16bf a0 = load_frag(aP +  0);
    v16bf p0 = load_frag(g0 +  0);
    v16bf q0 = load_frag(g1 +  0);
    v16bf r0 = load_frag(g2 +  0);
    v16bf s0 = load_frag(g3 +  0);
    v16bf a1 = load_frag(aP + 32);
    v16bf p1 = load_frag(g0 + 32);
    v16bf q1 = load_frag(g1 + 32);
    v16bf r1 = load_frag(g2 + 32);
    v16bf s1 = load_frag(g3 + 32);

#pragma unroll 4
    for (int k = 64; k < K; k += 32) {
        // prefetch stage k while computing stage k-64
        v16bf a2 = load_frag(aP + k);
        v16bf p2 = load_frag(g0 + k);
        v16bf q2 = load_frag(g1 + k);
        v16bf r2 = load_frag(g2 + k);
        v16bf s2 = load_frag(g3 + k);

        acc0 = wmma_bf16(a0, p0, acc0);
        acc1 = wmma_bf16(a0, q0, acc1);
        acc2 = wmma_bf16(a0, r0, acc2);
        acc3 = wmma_bf16(a0, s0, acc3);

        a0 = a1; p0 = p1; q0 = q1; r0 = r1; s0 = s1;
        a1 = a2; p1 = p2; q1 = q2; r1 = r2; s1 = s2;
    }
    // drain the two in-flight stages
    acc0 = wmma_bf16(a0, p0, acc0);
    acc1 = wmma_bf16(a0, q0, acc1);
    acc2 = wmma_bf16(a0, r0, acc2);
    acc3 = wmma_bf16(a0, s0, acc3);
    acc0 = wmma_bf16(a1, p1, acc0);
    acc1 = wmma_bf16(a1, q1, acc1);
    acc2 = wmma_bf16(a1, r1, acc2);
    acc3 = wmma_bf16(a1, s1, acc3);
}

// ---------------------------------------------------------------------------
// One timestep. One wave (32 threads) per workgroup; each wave owns a
// 16(b) x 16(n) output tile and accumulates all four gates in registers.
// grid = (HH/16, BB/16).
// ---------------------------------------------------------------------------
__global__ __launch_bounds__(32) void lstm_step(
    const bf16_t* __restrict__ Xbf,     // [T, B, IN] bf16
    const bf16_t* __restrict__ Wxb,     // [4H, IN]   bf16
    const bf16_t* __restrict__ Whb,     // [4H, H]    bf16
    const float*  __restrict__ bias,    // [4H]       f32
    const int*    __restrict__ mask,    // [T, B]
    const bf16_t* __restrict__ Cbf_in,  // [B, H] bf16 cell state (read)
    bf16_t*       __restrict__ Cbf_out, // [B, H] bf16 cell state (write)
    float*        __restrict__ Cst,     // [B, H] f32 master cell state
    float*        __restrict__ Hst,     // [B, H] f32 master hidden state
    float*        __restrict__ hiddens, // [T, B, H] output
    int t)
{
    const int lane  = threadIdx.x;
    const int ln    = lane & 15;          // N (or M for A) within tile
    const int koff  = (lane >> 4) * 8;    // K sub-offset per fragment layout
    const int nTile = blockIdx.x * 16;    // column tile within H
    const int bTile = blockIdx.y * 16;    // row tile within B
    const int rbase = (lane >> 4) * 8;    // C-layout row base for this lane
    const int n     = nTile + ln;

    // ---- Hoisted epilogue operands: latency hides under the GEMM ----------
    const float bi = bias[0 * HH + n];
    const float bf = bias[1 * HH + n];
    const float bo = bias[2 * HH + n];
    const float bg = bias[3 * HH + n];

    float c_old[8], h_old[8], mval[8];
#pragma unroll
    for (int r = 0; r < 8; ++r) {
        const int bm = bTile + rbase + r;
        const size_t idx = (size_t)bm * HH + n;
        c_old[r] = Cst[idx];
        h_old[r] = Hst[idx];
        mval[r]  = (float)mask[t * BB + bm];
    }

    v8f acc0 = {}, acc1 = {}, acc2 = {}, acc3 = {};

    // ---- Phase 1: input projection, K = IN --------------------------------
    gemm_phase<INN>(
        Xbf + ((size_t)t * BB + bTile + ln) * INN + koff,
        Wxb + ((size_t)(0 * HH + nTile + ln)) * INN + koff,
        Wxb + ((size_t)(1 * HH + nTile + ln)) * INN + koff,
        Wxb + ((size_t)(2 * HH + nTile + ln)) * INN + koff,
        Wxb + ((size_t)(3 * HH + nTile + ln)) * INN + koff,
        acc0, acc1, acc2, acc3);

    // ---- Phase 2: recurrent projection on CELL state, K = H ---------------
    gemm_phase<HH>(
        Cbf_in + (size_t)(bTile + ln) * HH + koff,
        Whb + ((size_t)(0 * HH + nTile + ln)) * HH + koff,
        Whb + ((size_t)(1 * HH + nTile + ln)) * HH + koff,
        Whb + ((size_t)(2 * HH + nTile + ln)) * HH + koff,
        Whb + ((size_t)(3 * HH + nTile + ln)) * HH + koff,
        acc0, acc1, acc2, acc3);

    // ---- Epilogue: gates + state update, all in registers -----------------
    // C layout: element r of the v8f <-> row M = r + 8*(lane>=16), col N = lane&15.
#pragma unroll
    for (int r = 0; r < 8; ++r) {
        const int bm = bTile + rbase + r;
        const float pi = acc0[r] + bi;
        const float pf = acc1[r] + bf;
        const float po = acc2[r] + bo;
        const float pg = acc3[r] + bg;

        const float ig = sigmoidf_fast(pi);
        const float fg = sigmoidf_fast(pf);
        const float og = sigmoidf_fast(po);
        const float gg = tanhf(pg);

        float c_new = fg * c_old[r] + ig * gg;
        float h_new = og * tanhf(c_new);

        const float m = mval[r];
        h_new = m * h_new + (1.0f - m) * h_old[r];
        c_new = m * c_new + (1.0f - m) * c_old[r];

        const size_t idx = (size_t)bm * HH + n;
        // Streaming (write-once) output: keep it out of L2 so the bf16
        // weights stay resident.
        __builtin_nontemporal_store(h_new, &hiddens[((size_t)t * BB + bm) * HH + n]);
        Hst[idx] = h_new;
        Cst[idx] = c_new;
        Cbf_out[idx] = (bf16_t)c_new;
    }
}

// ---------------------------------------------------------------------------
// Helpers
// ---------------------------------------------------------------------------
__global__ __launch_bounds__(256) void cvt_f32_bf16(
    const float* __restrict__ in, bf16_t* __restrict__ out, int n)
{
    int i = blockIdx.x * blockDim.x + threadIdx.x;
    if (i < n) out[i] = (bf16_t)in[i];
}

__global__ __launch_bounds__(256) void init_state(
    const float* __restrict__ h0, const float* __restrict__ c0,
    float* __restrict__ Hst, float* __restrict__ Cst,
    bf16_t* __restrict__ Cbf, int n)
{
    int i = blockIdx.x * blockDim.x + threadIdx.x;
    if (i < n) {
        float h = h0[i];
        float c = c0[i];
        Hst[i] = h;
        Cst[i] = c;
        Cbf[i] = (bf16_t)c;
    }
}

__global__ __launch_bounds__(256) void finalize_state(
    const float* __restrict__ Hst, const float* __restrict__ Cst,
    float* __restrict__ outh, float* __restrict__ outc, int n)
{
    int i = blockIdx.x * blockDim.x + threadIdx.x;
    if (i < n) {
        outh[i] = Hst[i];
        outc[i] = Cst[i];
    }
}

// ---------------------------------------------------------------------------
// Launch
// ---------------------------------------------------------------------------
extern "C" void kernel_launch(void* const* d_in, const int* in_sizes, int n_in,
                              void* d_out, int out_size, void* d_ws, size_t ws_size,
                              hipStream_t stream)
{
    (void)in_sizes; (void)n_in; (void)out_size; (void)ws_size;

    const float* X    = (const float*)d_in[0];   // [T,B,IN]
    const float* h0   = (const float*)d_in[1];   // [B,H]
    const float* c0   = (const float*)d_in[2];   // [B,H]
    const int*   mask = (const int*)  d_in[3];   // [T,B]
    const float* Wx   = (const float*)d_in[4];   // [4H,IN]
    const float* Wh   = (const float*)d_in[5];   // [4H,H]
    const float* bias = (const float*)d_in[6];   // [4H]
    float* out = (float*)d_out;

    const size_t nX  = (size_t)TT * BB * INN;    // 33,554,432
    const size_t nWx = (size_t)4 * HH * INN;     //  4,194,304
    const size_t nWh = (size_t)4 * HH * HH;      //  4,194,304
    const size_t nBH = (size_t)BB * HH;          //     65,536

    // Workspace layout (all 256B aligned): ~84.7 MB total.
    char* ws = (char*)d_ws;
    size_t off = 0;
    auto take = [&](size_t bytes) -> char* {
        char* p = ws + off;
        off += (bytes + 255) & ~(size_t)255;
        return p;
    };
    bf16_t* Xbf  = (bf16_t*)take(nX  * sizeof(bf16_t));
    bf16_t* Wxb  = (bf16_t*)take(nWx * sizeof(bf16_t));
    bf16_t* Whb  = (bf16_t*)take(nWh * sizeof(bf16_t));
    float*  Cst  = (float*) take(nBH * sizeof(float));
    float*  Hst  = (float*) take(nBH * sizeof(float));
    bf16_t* Cbf0 = (bf16_t*)take(nBH * sizeof(bf16_t));
    bf16_t* Cbf1 = (bf16_t*)take(nBH * sizeof(bf16_t));

    // 1) Convert operands to bf16 (weights stay resident in the 192MB L2).
    cvt_f32_bf16<<<(unsigned)((nX  + 255) / 256), 256, 0, stream>>>(X,  Xbf, (int)nX);
    cvt_f32_bf16<<<(unsigned)((nWx + 255) / 256), 256, 0, stream>>>(Wx, Wxb, (int)nWx);
    cvt_f32_bf16<<<(unsigned)((nWh + 255) / 256), 256, 0, stream>>>(Wh, Whb, (int)nWh);

    // 2) Initialize fp32 master state + bf16 cell buffer 0.
    init_state<<<(unsigned)((nBH + 255) / 256), 256, 0, stream>>>(
        h0, c0, Hst, Cst, Cbf0, (int)nBH);

    // 3) Sequential scan: one WMMA kernel per timestep, double-buffered c_bf16.
    dim3 grid(HH / 16, BB / 16);   // 64 x 4 = 256 single-wave workgroups
    for (int t = 0; t < TT; ++t) {
        const bf16_t* cin  = (t & 1) ? Cbf1 : Cbf0;
        bf16_t*       cout = (t & 1) ? Cbf0 : Cbf1;
        lstm_step<<<grid, 32, 0, stream>>>(
            Xbf, Wxb, Whb, bias, mask, cin, cout, Cst, Hst, out, t);
    }

    // 4) Emit h_last, c_last after the hiddens block.
    float* outh = out + (size_t)TT * BB * HH;
    float* outc = outh + nBH;
    finalize_state<<<(unsigned)((nBH + 255) / 256), 256, 0, stream>>>(
        Hst, Cst, outh, outc, (int)nBH);
}
